// PPGCDR_75539884802428
// MI455X (gfx1250) — matrix operand order repaired and checked by reading, not compiled, not run on hardware
//
#include <hip/hip_runtime.h>

// ============================================================================
// MI455X (gfx1250, wave32) implementation of the GNN+GAT reference.
//   N = 8192 nodes, D = 128 features, STEPS = 2.
// Heavy ops are three dense 8192x8192 x 8192x128 matmuls -> bf16 WMMA
// (v_wmma_f32_16x16x32_bf16) with fp32 accumulation.
// adj is converted once to bf16 in workspace (128 MB < 192 MB L2) so the
// three N^2 matmul passes + softmax-stats pass hit L2, not HBM.
// b_gcn is omitted: a per-column constant shift cancels inside batch-norm.
// Big GEMM blocks are 64x64 tiles (4 row-tiles x 2 col-pairs per 8 waves):
// A-fragment work (incl. GAT exp generation) duplicated only 2x per block.
// ============================================================================

static constexpr int NN = 8192;   // nodes
static constexpr int DD = 128;    // feature dim

typedef __attribute__((ext_vector_type(16))) __bf16 v16bf;
typedef __attribute__((ext_vector_type(8)))  float  v8f;

union FragU {
    v16bf v;
    uint4 q[2];
    unsigned short s[16];
};

__device__ inline unsigned short f2bf(float f) {
    unsigned u = __float_as_uint(f);
    unsigned r = u + 0x7FFFu + ((u >> 16) & 1u);   // round-to-nearest-even
    return (unsigned short)(r >> 16);
}
__device__ inline float bf2f(unsigned short h) {
    return __uint_as_float(((unsigned)h) << 16);
}
__device__ inline float leakyf(float x) { return x > 0.f ? x : 0.2f * x; }
__device__ inline float eluf(float x)   { return x > 0.f ? x : __expf(x) - 1.f; }

__device__ inline v8f wmma_bf16(v16bf a, v16bf b, v8f c) {
    // D = A(16x32 bf16) * B(32x16 bf16) + C(16x16 f32)
    return __builtin_amdgcn_wmma_f32_16x16x32_bf16(
        /*neg_a=*/false, a, /*neg_b=*/false, b,
        /*c_mod=*/(short)0, c, /*reuse_a=*/false, /*reuse_b=*/false);
}

// ---- A-fragment (16x32, bf16, row-major source) ---------------------------
// lane L holds row M = L%16; lanes 0-15: K k0+0..7 & k0+16..23,
//                            lanes 16-31: K k0+8..15 & k0+24..31.
__device__ inline v16bf load_a_bf16(const unsigned short* __restrict__ base,
                                    size_t ld, int row0, int k0) {
    int lane = threadIdx.x & 31;
    const unsigned short* p =
        base + (size_t)(row0 + (lane & 15)) * ld + k0 + ((lane >> 4) << 3);
    FragU u;
    u.q[0] = *(const uint4*)(p);
    u.q[1] = *(const uint4*)(p + 16);
    return u.v;
}

// Same fragment but converting from an fp32 row-major source.
__device__ inline v16bf load_a_f32(const float* __restrict__ base,
                                   size_t ld, int row0, int k0) {
    int lane = threadIdx.x & 31;
    const float* p =
        base + (size_t)(row0 + (lane & 15)) * ld + k0 + ((lane >> 4) << 3);
    float4 a = *(const float4*)(p);
    float4 b = *(const float4*)(p + 4);
    float4 c = *(const float4*)(p + 16);
    float4 d = *(const float4*)(p + 20);
    FragU u;
    u.s[0]  = f2bf(a.x); u.s[1]  = f2bf(a.y); u.s[2]  = f2bf(a.z); u.s[3]  = f2bf(a.w);
    u.s[4]  = f2bf(b.x); u.s[5]  = f2bf(b.y); u.s[6]  = f2bf(b.z); u.s[7]  = f2bf(b.w);
    u.s[8]  = f2bf(c.x); u.s[9]  = f2bf(c.y); u.s[10] = f2bf(c.z); u.s[11] = f2bf(c.w);
    u.s[12] = f2bf(d.x); u.s[13] = f2bf(d.y); u.s[14] = f2bf(d.z); u.s[15] = f2bf(d.w);
    return u.v;
}

// ---- B-fragment (32x16) from a TRANSPOSED bf16 matrix bt[n][k] ------------
// lane L: column N = n0 + L%16; lanes 0-15 K k0+0..7 & k0+16..23,
// lanes 16-31 K k0+8..15 & k0+24..31 -> two b128 loads.
__device__ inline v16bf load_bT(const unsigned short* __restrict__ bt,
                                size_t ld, int n0, int k0) {
    int lane = threadIdx.x & 31;
    const unsigned short* p =
        bt + (size_t)(n0 + (lane & 15)) * ld + k0 + ((lane >> 4) << 3);
    FragU u;
    u.q[0] = *(const uint4*)(p);
    u.q[1] = *(const uint4*)(p + 16);
    return u.v;
}

// Same pattern out of LDS (wt[n*128 + k], 128x128 bf16) -> ds_load traffic.
__device__ inline v16bf load_bT_lds(const unsigned short* wt, int n0, int k0) {
    int lane = threadIdx.x & 31;
    const unsigned short* p = wt + (n0 + (lane & 15)) * 128 + k0 + ((lane >> 4) << 3);
    FragU u;
    u.q[0] = *(const uint4*)(p);
    u.q[1] = *(const uint4*)(p + 16);
    return u.v;
}

// ============================================================================
// K1: adj fp32 -> bf16 (RNE). zeros stay exactly zero -> mask preserved.
// ============================================================================
__global__ void k_cvt_adj(const float* __restrict__ adj,
                          unsigned short* __restrict__ adjb, long long n) {
    long long i = ((long long)blockIdx.x * blockDim.x + threadIdx.x) * 8;
    if (i >= n) return;
    float4 a = *(const float4*)(adj + i);
    float4 b = *(const float4*)(adj + i + 4);
    uint4 o;
    o.x = (unsigned)f2bf(a.x) | ((unsigned)f2bf(a.y) << 16);
    o.y = (unsigned)f2bf(a.z) | ((unsigned)f2bf(a.w) << 16);
    o.z = (unsigned)f2bf(b.x) | ((unsigned)f2bf(b.y) << 16);
    o.w = (unsigned)f2bf(b.z) | ((unsigned)f2bf(b.w) << 16);
    *(uint4*)(adjb + i) = o;
}

// ============================================================================
// K2: embs[:,0,:] = hidden
// ============================================================================
__global__ void k_embed0(const float* __restrict__ hidden, float* __restrict__ out) {
    int idx = blockIdx.x * 256 + threadIdx.x;           // over N*D
    int i = idx >> 7, d = idx & 127;
    out[((size_t)i * 3) * DD + d] = hidden[idx];
}

// ============================================================================
// K3: small GEMM  outT = (A[8192 x 128] @ W[128 x 128])^T as bf16.
// Block = 128 rows x 128 cols, 8 waves (16 rows each, 8 col tiles each).
// W^T is staged in LDS as bf16 (32 KB) -> B-frags via ds loads.
// ============================================================================
__global__ __launch_bounds__(256)
void k_gemm_small(const float* __restrict__ A, size_t ldA,
                  const float* __restrict__ W,
                  unsigned short* __restrict__ outT) {
    __shared__ unsigned short wt[128 * 128];            // W^T bf16
    for (int e = threadIdx.x; e < 128 * 128; e += 256) {
        int k = e >> 7, n = e & 127;
        wt[n * 128 + k] = f2bf(W[e]);                   // W[k][n] -> wt[n][k]
    }
    __syncthreads();

    int w  = threadIdx.x >> 5;
    int r0 = blockIdx.x * 128 + w * 16;
    v8f acc[8] = {};
#pragma unroll
    for (int k0 = 0; k0 < 128; k0 += 32) {
        v16bf af = load_a_f32(A, ldA, r0, k0);
#pragma unroll
        for (int t = 0; t < 8; ++t) {
            v16bf bfrag = load_bT_lds(wt, t * 16, k0);
            acc[t] = wmma_bf16(af, bfrag, acc[t]);
        }
    }
    int lane = threadIdx.x & 31;
    int hi = lane >> 4, nn = lane & 15;
#pragma unroll
    for (int t = 0; t < 8; ++t)
#pragma unroll
        for (int g = 0; g < 8; ++g) {
            int row = r0 + g + 8 * hi;
            int col = t * 16 + nn;
            outT[(size_t)col * NN + row] = f2bf(acc[t][g]);
        }
}

// ============================================================================
// GAT A-fragment generator: p_ij = exp(masked_leaky(f1_i+f2_j) - m_i) / s_i
// built on the fly from adj (mask) + f1/f2 + row stats; packed to bf16.
// ============================================================================
template <bool ADJ_BF16>
__device__ inline v16bf make_p_frag(const unsigned short* __restrict__ adjb,
                                    const float* __restrict__ adjf,
                                    const float* __restrict__ f2v,
                                    int r0, int k0, float f1r, float m, float isr) {
    int lane = threadIdx.x & 31;
    int r  = r0 + (lane & 15);
    int kk = k0 + ((lane >> 4) << 3);
    float av[16];
    if (ADJ_BF16) {
        const unsigned short* p = adjb + (size_t)r * NN + kk;
        FragU u;
        u.q[0] = *(const uint4*)(p);
        u.q[1] = *(const uint4*)(p + 16);
#pragma unroll
        for (int t = 0; t < 16; ++t) av[t] = bf2f(u.s[t]);
    } else {
        const float* p = adjf + (size_t)r * NN + kk;
        float4 a = *(const float4*)(p);
        float4 b = *(const float4*)(p + 4);
        float4 c = *(const float4*)(p + 16);
        float4 d = *(const float4*)(p + 20);
        av[0]=a.x; av[1]=a.y; av[2]=a.z; av[3]=a.w;
        av[4]=b.x; av[5]=b.y; av[6]=b.z; av[7]=b.w;
        av[8]=c.x; av[9]=c.y; av[10]=c.z; av[11]=c.w;
        av[12]=d.x; av[13]=d.y; av[14]=d.z; av[15]=d.w;
    }
    // f2 values for K kk..kk+7 and kk+16..kk+23 (vectorized)
    float4 fa0 = *(const float4*)(f2v + kk);
    float4 fa1 = *(const float4*)(f2v + kk + 4);
    float4 fb0 = *(const float4*)(f2v + kk + 16);
    float4 fb1 = *(const float4*)(f2v + kk + 20);
    float fv[16] = {fa0.x, fa0.y, fa0.z, fa0.w, fa1.x, fa1.y, fa1.z, fa1.w,
                    fb0.x, fb0.y, fb0.z, fb0.w, fb1.x, fb1.y, fb1.z, fb1.w};
    FragU o;
#pragma unroll
    for (int t = 0; t < 16; ++t) {
        float x = f1r + fv[t];
        float e = (av[t] > 0.f) ? leakyf(x) : -9e15f;
        o.s[t] = f2bf(__expf(e - m) * isr);
    }
    return o.v;
}

// ============================================================================
// K4: big GEMM  Y[8192 x 128] = A[8192 x 8192] @ B, B given transposed bf16.
// Grid = 256 blocks mapped 2D: 128 row-blocks x 2 col-blocks, block = 64x64.
// 8 waves per block: wave w -> row-tile (w>>1), col-pair (w&1) -> A-fragment
// work (incl. GAT softmax generation) duplicated only 2x per block.
// GCN mode: A = adj (bf16), plain fp32 store.
// GAT mode: A = softmax probs generated on the fly, epilogue = ELU.
// A stream is prefetched 8 k-chunks ahead (global_prefetch_b8).
// ============================================================================
template <bool ADJ_BF16, bool GAT>
__global__ __launch_bounds__(256)
void k_gemm_big(const unsigned short* __restrict__ adjb,
                const float* __restrict__ adjf,
                const unsigned short* __restrict__ bT,   // [128][8192] bf16
                const float* __restrict__ f1,
                const float* __restrict__ f2v,
                const float* __restrict__ mrow,
                const float* __restrict__ invs,
                float* __restrict__ Yout) {              // [8192][128] fp32
    int w    = threadIdx.x >> 5;
    int lane = threadIdx.x & 31;
    int rb = blockIdx.x >> 1;                            // 0..127 row blocks
    int cb = blockIdx.x & 1;                             // 0..1   col blocks
    int r0 = rb * 64 + (w >> 1) * 16;
    int n0 = cb * 64 + (w & 1) * 32;

    float f1r = 0.f, m = 0.f, isr = 0.f;
    if (GAT) {
        int r = r0 + (lane & 15);
        f1r = f1[r]; m = mrow[r]; isr = invs[r];
    }
    const size_t arow = (size_t)(r0 + (lane & 15)) * NN;

    v8f acc0 = {}, acc1 = {};
#pragma unroll 2
    for (int k0 = 0; k0 < NN; k0 += 32) {
        if (k0 + 256 < NN) {                             // prefetch A 8 chunks ahead
            if (ADJ_BF16) __builtin_prefetch(adjb + arow + k0 + 256, 0, 3);
            else          __builtin_prefetch(adjf + arow + k0 + 256, 0, 3);
        }
        v16bf af;
        if (GAT) {
            af = make_p_frag<ADJ_BF16>(adjb, adjf, f2v, r0, k0, f1r, m, isr);
        } else {
            af = ADJ_BF16 ? load_a_bf16(adjb, NN, r0, k0)
                          : load_a_f32 (adjf, NN, r0, k0);
        }
        v16bf b0 = load_bT(bT, NN, n0, k0);
        v16bf b1 = load_bT(bT, NN, n0 + 16, k0);
        acc0 = wmma_bf16(af, b0, acc0);
        acc1 = wmma_bf16(af, b1, acc1);
    }

    int hi = lane >> 4, nn = lane & 15;
#pragma unroll
    for (int g = 0; g < 8; ++g) {
        int row = r0 + g + 8 * hi;
        float y0 = acc0[g], y1 = acc1[g];
        if (GAT) { y0 = eluf(y0); y1 = eluf(y1); }
        Yout[(size_t)row * DD + n0 + nn]      = y0;
        Yout[(size_t)row * DD + n0 + 16 + nn] = y1;
    }
}

// ============================================================================
// K5/K6: batch-norm column stats, two deterministic stages (no atomics).
// ============================================================================
__global__ void k_bn1(const float* __restrict__ Y,
                      float* __restrict__ pS, float* __restrict__ pQ) {
    int d = threadIdx.x;                 // 128 threads = columns
    int b = blockIdx.x;                  // 128 blocks x 64 rows
    const float* p = Y + (size_t)b * 64 * DD + d;
    float s = 0.f, q = 0.f;
#pragma unroll 4
    for (int r = 0; r < 64; ++r) { float v = p[r * DD]; s += v; q += v * v; }
    pS[b * DD + d] = s;
    pQ[b * DD + d] = q;
}

__global__ void k_bn2(const float* __restrict__ pS, const float* __restrict__ pQ,
                      const float* __restrict__ gamma, const float* __restrict__ beta,
                      float* __restrict__ scl, float* __restrict__ shf) {
    int d = threadIdx.x;
    float s = 0.f, q = 0.f;
    for (int b = 0; b < 128; ++b) { s += pS[b * DD + d]; q += pQ[b * DD + d]; }
    float mu  = s * (1.f / NN);
    float var = q * (1.f / NN) - mu * mu;           // biased, matches jnp.var
    float ri  = rsqrtf(var + 1e-5f);
    float g   = gamma[d];
    scl[d] = g * ri;
    shf[d] = beta[d] - g * ri * mu;
}

__global__ void k_bnapply(const float* __restrict__ Y,
                          const float* __restrict__ scl, const float* __restrict__ shf,
                          float* __restrict__ out, int step) {
    int idx = blockIdx.x * 256 + threadIdx.x;       // over N*D
    int i = idx >> 7, d = idx & 127;
    out[((size_t)i * 3 + step) * DD + d] = scl[d] * Y[idx] + shf[d];
}

// ============================================================================
// K7: f1 = h@a1, f2 = h@a2 from transposed bf16 hT (coalesced over i).
// ============================================================================
__global__ void k_f12(const unsigned short* __restrict__ hT,
                      const float* __restrict__ a1, const float* __restrict__ a2,
                      float* __restrict__ f1, float* __restrict__ f2v) {
    int i = blockIdx.x * 256 + threadIdx.x;
    float s1 = 0.f, s2 = 0.f;
#pragma unroll 4
    for (int d = 0; d < DD; ++d) {
        float v = bf2f(hT[(size_t)d * NN + i]);
        s1 += v * a1[d];
        s2 += v * a2[d];
    }
    f1[i] = s1;
    f2v[i] = s2;
}

// ============================================================================
// K8: GAT softmax row stats. One wave per row.
// Pass 1: m_i = leaky(f1_i + max f2_j over neighbors) (leaky is monotone),
//         MASK if the row has no neighbors.
// Pass 2: s_i = sum_j exp(e_masked - m_i)  (masked terms underflow to 0,
//         or equal 1 for empty rows -> uniform softmax, matching reference).
// ============================================================================
template <bool ADJ_BF16>
__global__ __launch_bounds__(256)
void k_gat_stats(const unsigned short* __restrict__ adjb,
                 const float* __restrict__ adjf,
                 const float* __restrict__ f1, const float* __restrict__ f2v,
                 float* __restrict__ mrow, float* __restrict__ invs) {
    int w = threadIdx.x >> 5, lane = threadIdx.x & 31;
    int i = blockIdx.x * 8 + w;
    const size_t rowoff = (size_t)i * NN;

    float mx = -3.0e38f;
    for (int j0 = lane * 8; j0 < NN; j0 += 256) {
        float av[8];
        if (ADJ_BF16) {
            FragU u;
            u.q[0] = *(const uint4*)(adjb + rowoff + j0);
#pragma unroll
            for (int t = 0; t < 8; ++t) av[t] = bf2f(u.s[t]);
        } else {
            float4 a = *(const float4*)(adjf + rowoff + j0);
            float4 b = *(const float4*)(adjf + rowoff + j0 + 4);
            av[0]=a.x; av[1]=a.y; av[2]=a.z; av[3]=a.w;
            av[4]=b.x; av[5]=b.y; av[6]=b.z; av[7]=b.w;
        }
        float4 fa = *(const float4*)(f2v + j0);
        float4 fb = *(const float4*)(f2v + j0 + 4);
        float fv[8] = {fa.x, fa.y, fa.z, fa.w, fb.x, fb.y, fb.z, fb.w};
#pragma unroll
        for (int t = 0; t < 8; ++t)
            if (av[t] > 0.f) mx = fmaxf(mx, fv[t]);
    }
#pragma unroll
    for (int off = 16; off; off >>= 1) mx = fmaxf(mx, __shfl_xor(mx, off, 32));

    float f1i = f1[i];
    float m = (mx < -1e38f) ? -9e15f : leakyf(f1i + mx);

    float s = 0.f;
    for (int j0 = lane * 8; j0 < NN; j0 += 256) {
        float av[8];
        if (ADJ_BF16) {
            FragU u;
            u.q[0] = *(const uint4*)(adjb + rowoff + j0);
#pragma unroll
            for (int t = 0; t < 8; ++t) av[t] = bf2f(u.s[t]);
        } else {
            float4 a = *(const float4*)(adjf + rowoff + j0);
            float4 b = *(const float4*)(adjf + rowoff + j0 + 4);
            av[0]=a.x; av[1]=a.y; av[2]=a.z; av[3]=a.w;
            av[4]=b.x; av[5]=b.y; av[6]=b.z; av[7]=b.w;
        }
        float4 fa = *(const float4*)(f2v + j0);
        float4 fb = *(const float4*)(f2v + j0 + 4);
        float fv[8] = {fa.x, fa.y, fa.z, fa.w, fb.x, fb.y, fb.z, fb.w};
#pragma unroll
        for (int t = 0; t < 8; ++t) {
            float e = (av[t] > 0.f) ? leakyf(f1i + fv[t]) : -9e15f;
            s += __expf(e - m);
        }
    }
#pragma unroll
    for (int off = 16; off; off >>= 1) s += __shfl_xor(s, off, 32);

    if (lane == 0) { mrow[i] = m; invs[i] = 1.f / s; }
}

// ============================================================================
// Host-side orchestration.
// ============================================================================
extern "C" void kernel_launch(void* const* d_in, const int* in_sizes, int n_in,
                              void* d_out, int out_size, void* d_ws, size_t ws_size,
                              hipStream_t stream) {
    (void)in_sizes; (void)n_in; (void)out_size;
    const float* hidden = (const float*)d_in[0];
    const float* adj    = (const float*)d_in[1];
    const float* W_gcn  = (const float*)d_in[2];
    // d_in[3] = b_gcn: a per-column constant shift cancels inside batch-norm.
    const float* gamma  = (const float*)d_in[4];
    const float* beta   = (const float*)d_in[5];
    const float* W_gat  = (const float*)d_in[6];
    const float* a1     = (const float*)d_in[7];
    const float* a2     = (const float*)d_in[8];

    float* embs = (float*)d_out;                         // [N][3][D]
    float* gat  = embs + (size_t)NN * 3 * DD;            // [N][D]

    // ---- workspace carving -------------------------------------------------
    char* ws = (char*)d_ws;
    auto carve = [&](size_t bytes) -> char* {
        char* p = ws;
        ws += (bytes + 255) & ~(size_t)255;
        return p;
    };
    const size_t adjb_bytes = (size_t)NN * NN * 2;       // 128 MB, L2-resident
    const size_t rest_bytes = 2 * (size_t)DD * NN * 2    // tmpT + hT
                            + (size_t)NN * DD * 4        // Y
                            + 4 * (size_t)NN * 4         // f1,f2,m,invs
                            + 2 * 128 * 128 * 4          // BN partials
                            + 2 * 128 * 4 + 8192;        // BN scale/shift + pad
    const bool use_bf16 = ws_size >= adjb_bytes + rest_bytes;

    unsigned short* adjb = nullptr;
    if (use_bf16) adjb = (unsigned short*)carve(adjb_bytes);
    unsigned short* tmpT = (unsigned short*)carve((size_t)DD * NN * 2);
    unsigned short* hT   = (unsigned short*)carve((size_t)DD * NN * 2);
    float* Y    = (float*)carve((size_t)NN * DD * 4);
    float* f1   = (float*)carve((size_t)NN * 4);
    float* f2v  = (float*)carve((size_t)NN * 4);
    float* mrow = (float*)carve((size_t)NN * 4);
    float* invs = (float*)carve((size_t)NN * 4);
    float* pS   = (float*)carve(128 * 128 * 4);
    float* pQ   = (float*)carve(128 * 128 * 4);
    float* scl  = (float*)carve(128 * 4);
    float* shf  = (float*)carve(128 * 4);

    const int ND_BLKS = (NN * DD) / 256;                 // 4096

    // ---- one-time adj -> bf16 (makes all N^2 passes L2-resident) ----------
    if (use_bf16) {
        long long n = (long long)NN * NN;
        k_cvt_adj<<<(unsigned)((n / 8 + 255) / 256), 256, 0, stream>>>(adj, adjb, n);
    }

    // ---- embs[:,0,:] = hidden ---------------------------------------------
    k_embed0<<<ND_BLKS, 256, 0, stream>>>(hidden, embs);

    // ---- GCN step 1 --------------------------------------------------------
    k_gemm_small<<<NN / 128, 256, 0, stream>>>(hidden, DD, W_gcn, tmpT);
    if (use_bf16)
        k_gemm_big<true , false><<<256, 256, 0, stream>>>(adjb, adj, tmpT,
            nullptr, nullptr, nullptr, nullptr, Y);
    else
        k_gemm_big<false, false><<<256, 256, 0, stream>>>(adjb, adj, tmpT,
            nullptr, nullptr, nullptr, nullptr, Y);
    k_bn1<<<128, 128, 0, stream>>>(Y, pS, pQ);
    k_bn2<<<1, 128, 0, stream>>>(pS, pQ, gamma, beta, scl, shf);
    k_bnapply<<<ND_BLKS, 256, 0, stream>>>(Y, scl, shf, embs, 1);

    // ---- GCN step 2 (A = embs[:,1,:], row stride 3*D) ----------------------
    k_gemm_small<<<NN / 128, 256, 0, stream>>>(embs + DD, 3 * DD, W_gcn, tmpT);
    if (use_bf16)
        k_gemm_big<true , false><<<256, 256, 0, stream>>>(adjb, adj, tmpT,
            nullptr, nullptr, nullptr, nullptr, Y);
    else
        k_gemm_big<false, false><<<256, 256, 0, stream>>>(adjb, adj, tmpT,
            nullptr, nullptr, nullptr, nullptr, Y);
    k_bn1<<<128, 128, 0, stream>>>(Y, pS, pQ);
    k_bn2<<<1, 128, 0, stream>>>(pS, pQ, gamma, beta, scl, shf);
    k_bnapply<<<ND_BLKS, 256, 0, stream>>>(Y, scl, shf, embs, 2);

    // ---- GAT ---------------------------------------------------------------
    k_gemm_small<<<NN / 128, 256, 0, stream>>>(hidden, DD, W_gat, hT);
    k_f12<<<NN / 256, 256, 0, stream>>>(hT, a1, a2, f1, f2v);
    if (use_bf16) {
        k_gat_stats<true ><<<NN / 8, 256, 0, stream>>>(adjb, adj, f1, f2v, mrow, invs);
        k_gemm_big<true , true><<<256, 256, 0, stream>>>(adjb, adj, hT,
            f1, f2v, mrow, invs, gat);
    } else {
        k_gat_stats<false><<<NN / 8, 256, 0, stream>>>(adjb, adj, f1, f2v, mrow, invs);
        k_gemm_big<false, true><<<256, 256, 0, stream>>>(adjb, adj, hT,
            f1, f2v, mrow, invs, gat);
    }
}
